// RsNetGraphConv_57509612093719
// MI455X (gfx1250) — matrix operand order repaired and checked by reading, not compiled, not run on hardware
//
#include <hip/hip_runtime.h>

// MI455X / gfx1250 fused RsNetGraphConv kernel.
// Phase A: 12 GEMMs (h_self, h_neigh, x) in split-f16 (hi+lo, 3 products) via
//          v_wmma_f32_16x16x32_f16  -> ~f32 accuracy at f16 matrix throughput.
//          Weight B-fragments pre-swizzled into d_ws by a tiny pre-kernel.
// Phase B: Aoff neighbor mix via v_wmma_f32_16x16x4_f32 (j=0..15),
//          ragged j=16 column / i=16 row on VALU.

typedef float    v2f  __attribute__((ext_vector_type(2)));
typedef float    v4f  __attribute__((ext_vector_type(4)));
typedef float    v8f  __attribute__((ext_vector_type(8)));
typedef _Float16 v4h  __attribute__((ext_vector_type(4)));
typedef _Float16 v8h  __attribute__((ext_vector_type(8)));
typedef _Float16 v16h __attribute__((ext_vector_type(16)));

#define J_DIM   17
#define F_DIM   64
#define TILE_B  16
#define ROWS    (TILE_B * J_DIM)     // 272
#define XSTRH   72                   // input planes row stride (halves; conflict-free b128)
#define OSTR    64                   // sOut row stride (floats)
#define NSTR    68                   // sNeigh row stride (floats)

// LDS half-offsets (f16 planes first)
#define X1H 0
#define X1L (ROWS*XSTRH)             // 19584
#define X2H (2*ROWS*XSTRH)
#define X2L (3*ROWS*XSTRH)
// LDS float-offsets
#define OU_F 39168                   // = 4*ROWS*XSTRH/2  (bytes match)
#define NE_F (OU_F + ROWS*OSTR)      // 56576
#define M_F  (NE_F + ROWS*NSTR)      // 75072
#define A_F  (M_F + 4*J_DIM*F_DIM)   // 79424
#define D_F  (A_F + 4*J_DIM*18)      // 80648
#define BI_F (D_F + 4*20)            // 80728
#define LDS_BYTES ((BI_F + 256)*4)   // 323,936 B <= 320KB

#define WS_BYTES (192 * 512 * (int)sizeof(_Float16))   // 196,608 B of pre-swizzled frags

__device__ inline v16h frag_cat(v8h lo, v8h hi) {
    return __builtin_shufflevector(lo, hi, 0,1,2,3,4,5,6,7,8,9,10,11,12,13,14,15);
}

// Pre-swizzled fragment load: frag fi laid out as [lane][16 halves] contiguous.
__device__ inline v16h ldfrag(const _Float16* __restrict__ wf, int mat, int nt, int t,
                              int plane, int lane) {
    size_t base = ((size_t)(((mat*4 + nt)*2 + t)*2 + plane))*512 + (size_t)lane*16;
    v8h lo = *(const v8h*)(wf + base);
    v8h hi = *(const v8h*)(wf + base + 8);
    return frag_cat(lo, hi);
}

// Fallback: build hi/lo B-fragments from f32 weights inline.
// B f16 32x16 layout: lane L -> N = L&15 ; halves e<8 -> K = 8*(L>>4)+e, e>=8 -> K = 16+8*(L>>4)+(e-8).
__device__ inline void build_frag(const float* __restrict__ src, int nt, int t, int lane,
                                  v16h& fh, v16h& fl) {
    int g = lane >> 4, o = nt*16 + (lane & 15);
#pragma unroll
    for (int e = 0; e < 16; ++e) {
        float w = src[(32*t + ((e < 8) ? (8*g + e) : (16 + 8*g + (e - 8))))*64 + o];
        _Float16 hi = (_Float16)w;
        fh[e] = hi;
        fl[e] = (_Float16)(w - (float)hi);
    }
}

// Pre-kernel: dump all 192 weight B-fragments (12 mats x 4 ntiles x 2 ksteps x hi/lo).
__global__ __launch_bounds__(256, 1)
void build_wfrags(const float* __restrict__ W, const float* __restrict__ W2,
                  _Float16* __restrict__ wf) {
    int fi    = blockIdx.x;            // ((mat*4+nt)*2+t)*2+plane
    int plane = fi & 1;
    int t     = (fi >> 1) & 1;
    int nt    = (fi >> 2) & 3;
    int mat   = fi >> 4;
    const float* src = (mat < 8) ? (W + (size_t)mat*4096) : (W2 + (size_t)(mat-8)*4096);
    for (int u = threadIdx.x; u < 512; u += 256) {
        int lane = u >> 4, e = u & 15;
        int g = lane >> 4, o = nt*16 + (lane & 15);
        float w = src[(32*t + ((e < 8) ? (8*g + e) : (16 + 8*g + (e - 8))))*64 + o];
        _Float16 hi = (_Float16)w;
        wf[(size_t)fi*512 + u] = (plane == 0) ? hi : (_Float16)(w - (float)hi);
    }
}

__global__ __launch_bounds__(256, 1)
void rsnet_fused(const float* __restrict__ in1,  const float* __restrict__ in2,
                 const float* __restrict__ W,    const float* __restrict__ W2,
                 const float* __restrict__ Mm,   const float* __restrict__ adjs,
                 const float* __restrict__ adj2, const float* __restrict__ bias,
                 const _Float16* __restrict__ wfrag, int useWs,
                 float* __restrict__ out)
{
    extern __shared__ float smem[];
    _Float16* smemh = (_Float16*)smem;
    const int tid  = threadIdx.x;
    const int lane = tid & 31;
    const int wave = tid >> 5;
    const int b0   = blockIdx.x * TILE_B;

    // ---------------- stage tile (split hi/lo) + small tensors ----------------
    {
        const v4f* g1 = (const v4f*)(in1 + (size_t)b0 * J_DIM * F_DIM);
        const v4f* g2 = (const v4f*)(in2 + (size_t)b0 * J_DIM * F_DIM);
        for (int i4 = tid; i4 < ROWS * 16; i4 += 256) {     // float4-coalesced
            int r = i4 >> 4, c4 = i4 & 15;
            v4f x1 = g1[i4], x2 = g2[i4];
            v4h h1, l1, h2, l2;
#pragma unroll
            for (int e = 0; e < 4; ++e) {
                _Float16 a = (_Float16)x1[e]; h1[e] = a; l1[e] = (_Float16)(x1[e] - (float)a);
                _Float16 b = (_Float16)x2[e]; h2[e] = b; l2[e] = (_Float16)(x2[e] - (float)b);
            }
            int hoff = r*XSTRH + c4*4;
            *(v4h*)&smemh[X1H + hoff] = h1;  *(v4h*)&smemh[X1L + hoff] = l1;
            *(v4h*)&smemh[X2H + hoff] = h2;  *(v4h*)&smemh[X2L + hoff] = l2;
        }
        for (int i = tid; i < 4*J_DIM*F_DIM; i += 256) smem[M_F + i] = Mm[i];
        smem[BI_F + tid] = bias[tid];
        for (int e = tid; e < 4*J_DIM*J_DIM; e += 256) {
            int k   = e / (J_DIM*J_DIM);
            int rem = e - k*(J_DIM*J_DIM);
            int i   = rem / J_DIM;
            int j   = rem - i*J_DIM;
            int et  = k*(J_DIM*J_DIM) + j*J_DIM + i;
            float v = 0.5f * ((adjs[e] + adj2[e]) + (adjs[et] + adj2[et]));
            if (i == j) { smem[A_F + (k*J_DIM + i)*18 + j] = 0.0f;
                          smem[D_F + k*20 + i] = v; }
            else        { smem[A_F + (k*J_DIM + i)*18 + j] = v; }
        }
    }
    __syncthreads();

    const int lrow  = lane & 15;
    const int g     = lane >> 4;
    const int lkoff = g * 2;
    const int g8    = g * 8;

    for (int k = 0; k < 4; ++k) {
        const float* Ws = W  + (size_t)(2*k)     * 4096;
        const float* Wn = W  + (size_t)(2*k + 1) * 4096;
        const float* Wx = W2 + (size_t)k         * 4096;

        // ---------------- Phase A: split-f16 GEMMs on v_wmma_f32_16x16x32_f16 ----------------
        for (int nt = 0; nt < 4; ++nt) {
            v16h bSh[2], bSl[2], bNh[2], bNl[2], bXh[2], bXl[2];
            if (useWs) {
#pragma unroll
                for (int t = 0; t < 2; ++t) {
                    bSh[t] = ldfrag(wfrag, 2*k,   nt, t, 0, lane);
                    bSl[t] = ldfrag(wfrag, 2*k,   nt, t, 1, lane);
                    bNh[t] = ldfrag(wfrag, 2*k+1, nt, t, 0, lane);
                    bNl[t] = ldfrag(wfrag, 2*k+1, nt, t, 1, lane);
                    bXh[t] = ldfrag(wfrag, 8+k,   nt, t, 0, lane);
                    bXl[t] = ldfrag(wfrag, 8+k,   nt, t, 1, lane);
                }
            } else {
#pragma unroll
                for (int t = 0; t < 2; ++t) {
                    build_frag(Ws, nt, t, lane, bSh[t], bSl[t]);
                    build_frag(Wn, nt, t, lane, bNh[t], bNl[t]);
                    build_frag(Wx, nt, t, lane, bXh[t], bXl[t]);
                }
            }
            for (int rt = wave; rt < 17; rt += 8) {        // 17 row tiles / 8 waves
                const int r0 = rt * 16;
                const int j0 = r0 % J_DIM;                  // wave-uniform
                const int rowb = (r0 + lrow) * XSTRH;
                v8f accS = {}; v8f accN = {}; v8f accX = {};
#pragma unroll
                for (int t = 0; t < 2; ++t) {
                    // A f16 16x32: lane -> M=L&15; runs K = 32t+8g+(0..7), 32t+16+8g+(0..7)
                    int c0 = rowb + 32*t + 8*g;
                    v16h a1h = frag_cat(*(const v8h*)&smemh[X1H + c0],
                                        *(const v8h*)&smemh[X1H + c0 + 16]);
                    v16h a1l = frag_cat(*(const v8h*)&smemh[X1L + c0],
                                        *(const v8h*)&smemh[X1L + c0 + 16]);
                    v16h a2h = frag_cat(*(const v8h*)&smemh[X2H + c0],
                                        *(const v8h*)&smemh[X2H + c0 + 16]);
                    v16h a2l = frag_cat(*(const v8h*)&smemh[X2L + c0],
                                        *(const v8h*)&smemh[X2L + c0 + 16]);
                    accS = __builtin_amdgcn_wmma_f32_16x16x32_f16(false, a1h, false, bSh[t], (short)0, accS, false, false);
                    accN = __builtin_amdgcn_wmma_f32_16x16x32_f16(false, a1h, false, bNh[t], (short)0, accN, false, false);
                    accX = __builtin_amdgcn_wmma_f32_16x16x32_f16(false, a2h, false, bXh[t], (short)0, accX, false, false);
                    accS = __builtin_amdgcn_wmma_f32_16x16x32_f16(false, a1l, false, bSh[t], (short)0, accS, false, false);
                    accN = __builtin_amdgcn_wmma_f32_16x16x32_f16(false, a1l, false, bNh[t], (short)0, accN, false, false);
                    accX = __builtin_amdgcn_wmma_f32_16x16x32_f16(false, a2l, false, bXh[t], (short)0, accX, false, false);
                    accS = __builtin_amdgcn_wmma_f32_16x16x32_f16(false, a1h, false, bSl[t], (short)0, accS, false, false);
                    accN = __builtin_amdgcn_wmma_f32_16x16x32_f16(false, a1h, false, bNl[t], (short)0, accN, false, false);
                    accX = __builtin_amdgcn_wmma_f32_16x16x32_f16(false, a2h, false, bXl[t], (short)0, accX, false, false);
                }
#pragma unroll
                for (int v = 0; v < 8; ++v) {               // D: row = v+8g, col = lrow
                    int r  = r0 + v + g8;
                    int oo = nt*16 + lrow;
                    int jj = j0 + v + g8;  jj -= (jj >= J_DIM) ? J_DIM : 0;
                    float mval = smem[M_F + (k*J_DIM + jj)*F_DIM + oo];
                    smem[NE_F + r*NSTR + oo] = accN[v] * mval;
                    smem[OU_F + r*OSTR + oo] =
                          accS[v] * mval * smem[D_F + k*20 + jj]
                        + accX[v] + smem[BI_F + k*F_DIM + oo];
                }
            }
        }
        __syncthreads();

        // ------- Phase B: neighbor mix on f32 x4 WMMA -------
        {
            v2f aA[4];
#pragma unroll
            for (int t = 0; t < 4; ++t)
                aA[t] = *(const v2f*)&smem[A_F + (k*J_DIM + lrow)*18 + t*4 + lkoff];
            for (int bb = wave; bb < TILE_B; bb += 8) {
                const int rb = bb * J_DIM;
#pragma unroll
                for (int nt = 0; nt < 4; ++nt) {
                    const int col = nt*16 + lrow;
                    v8f acc = {};
#pragma unroll
                    for (int t = 0; t < 4; ++t) {
                        int c0 = t*4 + lkoff;
                        v2f bB;
                        bB.x = smem[NE_F + (rb + c0    )*NSTR + col];
                        bB.y = smem[NE_F + (rb + c0 + 1)*NSTR + col];
                        acc = __builtin_amdgcn_wmma_f32_16x16x4_f32(false, aA[t], false, bB, (short)0, acc, false, false);
                    }
                    float n16 = smem[NE_F + (rb + 16)*NSTR + col];
#pragma unroll
                    for (int v = 0; v < 8; ++v) {
                        int i = v + g8;
                        float val = acc[v]
                                  + smem[OU_F + (rb + i)*OSTR + col]
                                  + smem[A_F + (k*J_DIM + i)*18 + 16] * n16;
                        size_t off = ((size_t)(b0 + bb)*J_DIM + i)*256 + (size_t)k*64 + col;
                        out[off] = val;
                    }
                }
#pragma unroll
                for (int h = 0; h < 2; ++h) {               // ragged row i = 16
                    int o = lane + h*32;
                    float acc0 = smem[OU_F + (rb + 16)*OSTR + o];
#pragma unroll
                    for (int j = 0; j < J_DIM; ++j)
                        acc0 += smem[A_F + (k*J_DIM + 16)*18 + j]
                              * smem[NE_F + (rb + j)*NSTR + o];
                    out[((size_t)(b0 + bb)*J_DIM + 16)*256 + (size_t)k*64 + o] = acc0;
                }
            }
        }
        __syncthreads();
    }
}

extern "C" void kernel_launch(void* const* d_in, const int* in_sizes, int n_in,
                              void* d_out, int out_size, void* d_ws, size_t ws_size,
                              hipStream_t stream) {
    const float* in1  = (const float*)d_in[0];
    const float* in2  = (const float*)d_in[1];
    const float* W    = (const float*)d_in[2];
    const float* W2   = (const float*)d_in[3];
    const float* Mm   = (const float*)d_in[4];
    const float* adjs = (const float*)d_in[5];
    const float* adj2 = (const float*)d_in[6];
    const float* bias = (const float*)d_in[7];
    float* out = (float*)d_out;

    int useWs = (ws_size >= (size_t)WS_BYTES) ? 1 : 0;
    _Float16* wf = (_Float16*)d_ws;
    if (useWs)
        hipLaunchKernelGGL(build_wfrags, dim3(192), dim3(256), 0, stream, W, W2, wf);

    int Bv   = in_sizes[0] / (J_DIM * F_DIM);   // 8192
    int grid = Bv / TILE_B;                      // 512 workgroups
    hipLaunchKernelGGL(rsnet_fused, dim3(grid), dim3(256), LDS_BYTES, stream,
                       in1, in2, W, W2, Mm, adjs, adj2, bias, wf, useWs, out);
}
